// CausalSelfAttention_25555055411690
// MI455X (gfx1250) — compile-verified, matrix operand order checked by reading, not run on hardware
//
#include <hip/hip_runtime.h>
#include <hip/hip_bf16.h>

// ---------------------------------------------------------------------------
// Causal self-attention for MI455X (gfx1250, wave32, WMMA).
//   B=4, N=2048, D_IN=D_OUT=1024, fp32 in/out, f16 WMMA w/ f32 accumulate.
// Pipeline:
//   1) cvt_x:  x f32 -> xh f16
//   2) cvt_w:  W[k][n] f32 -> Wt[n][k] f16  (x3, so B-operand = row-major B^T)
//   3) qkv_gemm: Qh,Kh = xh@W (f16, [B,N,D]); Vt = (xh@Wv)^T (f16, [B,D,N])
//      -- 2x-unrolled ping-pong double buffering (no register-copy shift)
//   4) attn: flash-style online-softmax attention, f32 out
//      -- Q tile staged to LDS with GLOBAL_LOAD_ASYNC_TO_LDS (ASYNCcnt path)
// ---------------------------------------------------------------------------

typedef __attribute__((ext_vector_type(16))) _Float16 v16h;
typedef __attribute__((ext_vector_type(8)))  _Float16 v8h;
typedef __attribute__((ext_vector_type(8)))  float    v8f;

#define BDIM 4
#define NDIM 2048
#define DDIM 1024

#ifndef __has_builtin
#define __has_builtin(x) 0
#endif

#if __has_builtin(__builtin_amdgcn_global_load_async_to_lds_b128)
#define ASYNC_LDS 1
typedef int v4i_t __attribute__((__vector_size__(4 * sizeof(int))));
typedef __attribute__((address_space(1))) v4i_t GV4;  // global int4
typedef __attribute__((address_space(3))) v4i_t LV4;  // LDS int4
#else
#define ASYNC_LDS 0
#endif

__device__ __forceinline__ void wait_asynccnt0() {
#if __has_builtin(__builtin_amdgcn_s_wait_asynccnt)
  __builtin_amdgcn_s_wait_asynccnt(0);
#else
  asm volatile("s_wait_asynccnt 0x0" ::: "memory");
#endif
}

// Load a 16x32 f16 fragment (A-operand layout; also used for row-major B^T).
// Per CDNA5 ISA 16-bit A layout: lanes 0-15 -> rows 0-15 holding K 0..7 &
// 16..23; lanes 16-31 -> rows 0-15 holding K 8..15 & 24..31. Each lane reads
// two contiguous 16B chunks.
__device__ __forceinline__ v16h load_frag(const _Float16* base, int stride) {
  const int lane = threadIdx.x & 31;
  const int r    = lane & 15;
  const int kb   = (lane < 16) ? 0 : 8;
  const _Float16* p = base + (size_t)r * (size_t)stride + kb;
  v8h lo = *reinterpret_cast<const v8h*>(p);
  v8h hi = *reinterpret_cast<const v8h*>(p + 16);
  v16h f;
#pragma unroll
  for (int i = 0; i < 8; ++i) { f[i] = lo[i]; f[i + 8] = hi[i]; }
  return f;
}

__device__ __forceinline__ v8f wmma_f16(v16h a, v16h b, v8f c) {
  return __builtin_amdgcn_wmma_f32_16x16x32_f16(false, a, false, b,
                                                (short)0, c, false, false);
}

// ---------------------------------------------------------------------------
// Kernel 1: f32 -> f16 elementwise (vectorized x4)
__global__ __launch_bounds__(256) void cvt_x_kernel(const float* __restrict__ x,
                                                    _Float16* __restrict__ xh,
                                                    int n4) {
  int i = blockIdx.x * blockDim.x + threadIdx.x;
  if (i >= n4) return;
  float4 v = reinterpret_cast<const float4*>(x)[i];
  union { _Float16 h[4]; uint2 u; } o;
  o.h[0] = (_Float16)v.x; o.h[1] = (_Float16)v.y;
  o.h[2] = (_Float16)v.z; o.h[3] = (_Float16)v.w;
  reinterpret_cast<uint2*>(xh)[i] = o.u;
}

// Kernel 2: transpose+convert the three weight matrices: Wt[z][n][k] = W_z[k][n]
__global__ __launch_bounds__(256) void cvt_w_kernel(const float* __restrict__ W0,
                                                    const float* __restrict__ W1,
                                                    const float* __restrict__ W2,
                                                    _Float16* __restrict__ Wt) {
  int i = blockIdx.x * blockDim.x + threadIdx.x;
  if (i >= 3 * DDIM * DDIM) return;
  int z   = i >> 20;
  int rem = i & ((DDIM * DDIM) - 1);
  int n   = rem >> 10;
  int k   = rem & (DDIM - 1);
  const float* W = (z == 0) ? W0 : (z == 1) ? W1 : W2;
  Wt[i] = (_Float16)W[k * DDIM + n];
}

// ---------------------------------------------------------------------------
// Kernel 3: QKV projection GEMM.  M = B*N = 8192, Nc = 1024, K = 1024.
// grid = (M/64, Nc/64, 3), block = 128 (4 waves). Wave w computes rows
// [m0+16w, m0+16w+16) x cols [n0, n0+64). k-loop unrolled 2x into an explicit
// ping-pong: stage A loads chunk c+1 while computing chunk c from buffer 0,
// stage B loads chunk c+2 while computing chunk c+1 from buffer 1 -- no
// buffer-shift register copies, loads always a full stage ahead of the WMMAs.
__global__ __launch_bounds__(128) void qkv_gemm_kernel(
    const _Float16* __restrict__ xh, const _Float16* __restrict__ Wt,
    _Float16* __restrict__ Qh, _Float16* __restrict__ Kh,
    _Float16* __restrict__ Vt) {
  const int wid  = threadIdx.x >> 5;
  const int lane = threadIdx.x & 31;
  const int z    = blockIdx.z;
  const int m0   = blockIdx.x * 64 + wid * 16;
  const int n0   = blockIdx.y * 64;
  const _Float16* Wz    = Wt + (size_t)z * DDIM * DDIM;
  const _Float16* Abase = xh + (size_t)m0 * DDIM;

  v8f acc[4];
#pragma unroll
  for (int t = 0; t < 4; ++t) acc[t] = (v8f)0.0f;

  // prologue: buffer 0 <- chunk at kc = 0
  v16h a0 = load_frag(Abase, DDIM);
  v16h b0[4];
#pragma unroll
  for (int t = 0; t < 4; ++t)
    b0[t] = load_frag(Wz + (size_t)(n0 + 16 * t) * DDIM, DDIM);

  int kc = 0;
  for (; kc + 64 < DDIM; kc += 64) {
    __builtin_prefetch(Abase + kc + 128, 0, 3);
    // stage A: load chunk kc+32 into buffer 1, compute chunk kc from buffer 0
    v16h a1 = load_frag(Abase + kc + 32, DDIM);
    v16h b1[4];
#pragma unroll
    for (int t = 0; t < 4; ++t)
      b1[t] = load_frag(Wz + (size_t)(n0 + 16 * t) * DDIM + kc + 32, DDIM);
#pragma unroll
    for (int t = 0; t < 4; ++t) acc[t] = wmma_f16(a0, b0[t], acc[t]);
    // stage B: load chunk kc+64 into buffer 0, compute chunk kc+32 from buf 1
    a0 = load_frag(Abase + kc + 64, DDIM);
#pragma unroll
    for (int t = 0; t < 4; ++t)
      b0[t] = load_frag(Wz + (size_t)(n0 + 16 * t) * DDIM + kc + 64, DDIM);
#pragma unroll
    for (int t = 0; t < 4; ++t) acc[t] = wmma_f16(a1, b1[t], acc[t]);
  }
  // epilogue: kc == DDIM-64; buffer 0 holds chunk kc; one chunk remains
  {
    v16h a1 = load_frag(Abase + kc + 32, DDIM);
    v16h b1[4];
#pragma unroll
    for (int t = 0; t < 4; ++t)
      b1[t] = load_frag(Wz + (size_t)(n0 + 16 * t) * DDIM + kc + 32, DDIM);
#pragma unroll
    for (int t = 0; t < 4; ++t) acc[t] = wmma_f16(a0, b0[t], acc[t]);
#pragma unroll
    for (int t = 0; t < 4; ++t) acc[t] = wmma_f16(a1, b1[t], acc[t]);
  }

  const int colb  = lane & 15;
  const int rbase = (lane < 16) ? 0 : 8;
#pragma unroll
  for (int t = 0; t < 4; ++t) {
#pragma unroll
    for (int j = 0; j < 8; ++j) {
      int mg = m0 + rbase + j;
      int n  = n0 + 16 * t + colb;
      _Float16 hv = (_Float16)acc[t][j];
      if (z == 0) {
        Qh[(size_t)mg * DDIM + n] = hv;
      } else if (z == 1) {
        Kh[(size_t)mg * DDIM + n] = hv;
      } else {  // V stored transposed: Vt[b][n][token]
        int b   = mg >> 11;
        int tok = mg & (NDIM - 1);
        Vt[((size_t)b * DDIM + n) * NDIM + tok] = hv;
      }
    }
  }
}

// ---------------------------------------------------------------------------
// Kernel 4: flash attention. grid = (N/16, B), block = 256 (8 waves).
// Block owns 16 query rows; wave w owns context d-slice [128w, 128w+128).
// Per 32-key block: S reduction over d split across waves (8 WMMA each),
// partials summed via LDS, redundant online softmax per wave, wave0 publishes
// P(f16) to LDS, each wave does 8 PV WMMAs into its private accumulators.
__global__ __launch_bounds__(256) void attn_kernel(
    const _Float16* __restrict__ Qh, const _Float16* __restrict__ Kh,
    const _Float16* __restrict__ Vt, float* __restrict__ out) {
  const int b    = blockIdx.y;
  const int q0   = blockIdx.x * 16;
  const int tid  = threadIdx.x;
  const int w    = tid >> 5;
  const int lane = tid & 31;

  __shared__ _Float16 Qlds[16 * DDIM];        // 32 KB
  __shared__ float    Sred[8][2][8][32];      // 16 KB
  __shared__ _Float16 Plds[16][32];           // 1 KB

  // Stage the 16x1024 Q tile into LDS (16B chunks, all 256 threads).
  {
    const _Float16* Qrow = Qh + ((size_t)b * NDIM + q0) * DDIM;
#if ASYNC_LDS
    for (int i = tid; i < 16 * DDIM / 8; i += 256) {
      int row  = i >> 7;
      int col8 = (i & 127) * 8;
      __builtin_amdgcn_global_load_async_to_lds_b128(
          (GV4*)(Qrow + (size_t)row * DDIM + col8),
          (LV4*)(&Qlds[row * DDIM + col8]), 0, 0);
    }
    wait_asynccnt0();
#else
    for (int i = tid; i < 16 * DDIM / 8; i += 256) {
      int row  = i >> 7;
      int col8 = (i & 127) * 8;
      *reinterpret_cast<uint4*>(&Qlds[row * DDIM + col8]) =
          *reinterpret_cast<const uint4*>(&Qrow[(size_t)row * DDIM + col8]);
    }
#endif
  }
  __syncthreads();

  const int colb  = lane & 15;
  const int rbase = (lane < 16) ? 0 : 8;
  const float inv_scale = 0.03125f;  // 1/sqrt(1024)

  float mrow[8], lrow[8];
  v8f acc[8];
#pragma unroll
  for (int j = 0; j < 8; ++j) { mrow[j] = -3.0e38f; lrow[j] = 0.0f; }
#pragma unroll
  for (int t = 0; t < 8; ++t) acc[t] = (v8f)0.0f;

  const int nblocks = (q0 + 47) >> 5;  // key blocks of 32, causal bound
  for (int jb = 0; jb < nblocks; ++jb) {
    const int j0 = jb << 5;

    // ---- partial S over this wave's 128-wide d slice (pipelined) ----
    const _Float16* Kb0 = Kh + ((size_t)b * NDIM + j0) * DDIM;
    const _Float16* Kb1 = Kh + ((size_t)b * NDIM + j0 + 16) * DDIM;
    const int kc0 = w * 128;
    __builtin_prefetch(Kh + ((size_t)b * NDIM + j0 + 32) * DDIM + kc0, 0, 3);

    v8f s0 = (v8f)0.0f, s1 = (v8f)0.0f;
    v16h aq = load_frag(&Qlds[kc0], DDIM);
    v16h k0 = load_frag(Kb0 + kc0, DDIM);
    v16h k1 = load_frag(Kb1 + kc0, DDIM);
#pragma unroll
    for (int c = 0; c < 4; ++c) {
      v16h an = aq, k0n = k0, k1n = k1;
      if (c < 3) {
        const int kn = kc0 + (c + 1) * 32;
        an  = load_frag(&Qlds[kn], DDIM);
        k0n = load_frag(Kb0 + kn, DDIM);
        k1n = load_frag(Kb1 + kn, DDIM);
      }
      s0 = wmma_f16(aq, k0, s0);
      s1 = wmma_f16(aq, k1, s1);
      aq = an; k0 = k0n; k1 = k1n;
    }
#pragma unroll
    for (int j = 0; j < 8; ++j) {
      Sred[w][0][j][lane] = s0[j];
      Sred[w][1][j][lane] = s1[j];
    }
    __syncthreads();

    // ---- cross-wave reduce (every wave keeps a full copy) ----
#pragma unroll
    for (int j = 0; j < 8; ++j) {
      float t0 = 0.f, t1 = 0.f;
#pragma unroll
      for (int ww = 0; ww < 8; ++ww) {
        t0 += Sred[ww][0][j][lane];
        t1 += Sred[ww][1][j][lane];
      }
      s0[j] = t0; s1[j] = t1;
    }

    // ---- online softmax (redundant per wave; row reduce via shfl_xor) ----
    float al[8];
#pragma unroll
    for (int j = 0; j < 8; ++j) {
      const int row = q0 + rbase + j;
      const int c0  = j0 + colb;
      const int c1  = c0 + 16;
      float x0 = (c0 <= row) ? s0[j] * inv_scale : -3.0e38f;
      float x1 = (c1 <= row) ? s1[j] * inv_scale : -3.0e38f;
      float rm = fmaxf(x0, x1);
#pragma unroll
      for (int off = 8; off; off >>= 1) rm = fmaxf(rm, __shfl_xor(rm, off, 32));
      float mn = fmaxf(mrow[j], rm);
      float a_ = __expf(mrow[j] - mn);
      float p0 = __expf(x0 - mn);
      float p1 = __expf(x1 - mn);
      float rs = p0 + p1;
#pragma unroll
      for (int off = 8; off; off >>= 1) rs += __shfl_xor(rs, off, 32);
      lrow[j] = lrow[j] * a_ + rs;
      mrow[j] = mn;
      al[j]   = a_;
      if (w == 0) {  // publish P (f16) once; all waves computed identical P
        Plds[rbase + j][colb]      = (_Float16)p0;
        Plds[rbase + j][colb + 16] = (_Float16)p1;
      }
    }
    __syncthreads();

    // ---- PV update on this wave's 16x128 context slice (pipelined) ----
    const _Float16* Vb = Vt + ((size_t)b * DDIM + w * 128) * NDIM + j0;
    v16h pf = load_frag(&Plds[0][0], 32);
    v16h bv = load_frag(Vb, NDIM);
#pragma unroll
    for (int t = 0; t < 8; ++t) {
      v16h bvn = bv;
      if (t < 7) bvn = load_frag(Vb + (size_t)(t + 1) * 16 * NDIM, NDIM);
#pragma unroll
      for (int j = 0; j < 8; ++j) acc[t][j] *= al[j];
      acc[t] = wmma_f16(pf, bv, acc[t]);
      bv = bvn;
    }
  }

  // ---- normalize and store f32 output ----
#pragma unroll
  for (int t = 0; t < 8; ++t) {
#pragma unroll
    for (int j = 0; j < 8; ++j) {
      const int row = q0 + rbase + j;
      const int d   = w * 128 + t * 16 + colb;
      out[((size_t)b * NDIM + row) * DDIM + d] = acc[t][j] / lrow[j];
    }
  }
}

// ---------------------------------------------------------------------------
extern "C" void kernel_launch(void* const* d_in, const int* in_sizes, int n_in,
                              void* d_out, int out_size, void* d_ws,
                              size_t ws_size, hipStream_t stream) {
  (void)in_sizes; (void)n_in; (void)out_size; (void)ws_size;
  const float* x  = (const float*)d_in[0];
  const float* Wq = (const float*)d_in[1];
  const float* Wk = (const float*)d_in[2];
  const float* Wv = (const float*)d_in[3];
  float* out = (float*)d_out;

  const size_t elems = (size_t)BDIM * NDIM * DDIM;   // 8,388,608
  char* ws = (char*)d_ws;
  size_t off = 0;
  _Float16* xh = (_Float16*)(ws + off); off += elems * sizeof(_Float16);
  _Float16* Wt = (_Float16*)(ws + off); off += (size_t)3 * DDIM * DDIM * sizeof(_Float16);
  _Float16* Qh = (_Float16*)(ws + off); off += elems * sizeof(_Float16);
  _Float16* Kh = (_Float16*)(ws + off); off += elems * sizeof(_Float16);
  _Float16* Vt = (_Float16*)(ws + off); off += elems * sizeof(_Float16);

  // 1) convert x (vectorized by 4)
  {
    int n4 = (int)(elems / 4);
    cvt_x_kernel<<<(n4 + 255) / 256, 256, 0, stream>>>(x, xh, n4);
  }
  // 2) transpose+convert weights
  {
    int n = 3 * DDIM * DDIM;
    cvt_w_kernel<<<(n + 255) / 256, 256, 0, stream>>>(Wq, Wk, Wv, Wt);
  }
  // 3) QKV projections
  {
    dim3 grid((BDIM * NDIM) / 64, DDIM / 64, 3);
    qkv_gemm_kernel<<<grid, 128, 0, stream>>>(xh, Wt, Qh, Kh, Vt);
  }
  // 4) flash attention
  {
    dim3 grid(NDIM / 16, BDIM);
    attn_kernel<<<grid, 256, 0, stream>>>(Qh, Kh, Vt, out);
  }
}
